// GSMNet_455266533750
// MI455X (gfx1250) — matrix operand order repaired and checked by reading, not compiled, not run on hardware
//
#include <hip/hip_runtime.h>
#include <hip/hip_bf16.h>

typedef __bf16 gbf;
typedef gbf   v16bf __attribute__((ext_vector_type(16)));
typedef gbf   v4bf  __attribute__((ext_vector_type(4)));
typedef float v8f   __attribute__((ext_vector_type(8)));
typedef float v4f   __attribute__((ext_vector_type(4)));

#define HD 256
#define TMB 64              // edges per block
#define CUTOFF_F 5.0f
#define M_PI_F 3.14159265358979323846f

// packed-weight ktile bases (rows/32) in the bf16 weight pack
#define KT_EDGE 0
#define KT_LEN  8
#define KT_ANG  16
#define KT_UP1  24
#define KT_UP2  48
#define KT_GATE 56
#define KT_F1   72
#define KT_F2   96
#define KT_M1   104
#define KT_M2   128
#define KT_TOT  136

__device__ __forceinline__ float gsm_sigmoid(float x){ return 1.0f / (1.0f + __expf(-x)); }
__device__ __forceinline__ float gsm_silu(float x){ return x * gsm_sigmoid(x); }

// Offset of element (m, k) inside an A-layout LDS buffer laid out [mt][kt][lane][16]
// per the CDNA5 16-bit A-matrix 16x32 lane layout (lanes = M, with K split by bit3/bit4).
// For k aligned to 4, elements k..k+3 are contiguous in the same lane slot.
__device__ __forceinline__ int gsm_a_off(int m, int k){
  int kt   = k >> 5;
  int k32  = k & 31;
  int lane = (m & 15) | (((k32 >> 3) & 1) << 4);
  int e    = (k32 & 7) | ((k32 & 16) >> 1);
  return (((m >> 4) * 8 + kt) * 32 + lane) * 16 + e;
}

__device__ __forceinline__ void gsm_init_acc(v8f acc[4][2], const float* bias, int colBase, int lane){
  float b0 = bias[colBase + (lane & 15)];
  float b1 = bias[colBase + 16 + (lane & 15)];
  #pragma unroll
  for (int mt = 0; mt < 4; ++mt){
    #pragma unroll
    for (int r = 0; r < 8; ++r){ acc[mt][0][r] = b0; acc[mt][1][r] = b1; }
  }
}

// acc[mt][nt] += A(mt) x B(nt) over nseg consecutive 256-deep K segments.
// All 4 A tiles are loaded before issuing the 8 WMMAs so the ds_loads clause up
// and the dscnt wait is paid once per k-tile instead of once per m-tile.
__device__ __forceinline__ void gsm_gemm(v8f acc[4][2], const gbf* const* abufs, int nseg,
                                         int ktgBase, const gbf* __restrict__ wp,
                                         int lane, int colBase){
  const int nt0 = colBase >> 4;
  for (int s = 0; s < nseg; ++s){
    const gbf* ab = abufs[s];
    for (int kt = 0; kt < 8; ++kt){
      const int ktg = ktgBase + s*8 + kt;
      const v16bf b0 = *(const v16bf*)(wp + (size_t)(ktg*16 + nt0    ) * 512 + lane*16);
      const v16bf b1 = *(const v16bf*)(wp + (size_t)(ktg*16 + nt0 + 1) * 512 + lane*16);
      v16bf a[4];
      #pragma unroll
      for (int mt = 0; mt < 4; ++mt)
        a[mt] = *(const v16bf*)(ab + ((mt*8 + kt)*32 + lane)*16);
      #pragma unroll
      for (int mt = 0; mt < 4; ++mt){
        acc[mt][0] = __builtin_amdgcn_wmma_f32_16x16x32_bf16(false, a[mt], false, b0,
                        (short)0, acc[mt][0], false, false);
        acc[mt][1] = __builtin_amdgcn_wmma_f32_16x16x32_bf16(false, a[mt], false, b1,
                        (short)0, acc[mt][1], false, false);
      }
    }
  }
}

// C-layout: lanes 0-15 -> M=r, lanes 16-31 -> M=r+8 ; N = lane&15 within an n-tile.
__device__ __forceinline__ void gsm_store_f32(const v8f acc[4][2], float* f, int lane, int colBase){
  const int mhi = (lane >> 4) & 1, nl = lane & 15;
  #pragma unroll
  for (int mt = 0; mt < 4; ++mt)
    #pragma unroll
    for (int nt = 0; nt < 2; ++nt){
      const int col = colBase + nt*16 + nl;
      #pragma unroll
      for (int r = 0; r < 8; ++r)
        f[(mt*16 + r + mhi*8)*HD + col] = acc[mt][nt][r];
    }
}

__device__ __forceinline__ void gsm_store_bf16(const v8f acc[4][2], gbf* buf, int lane,
                                               int colBase, int do_silu){
  const int mhi = (lane >> 4) & 1, nl = lane & 15;
  #pragma unroll
  for (int mt = 0; mt < 4; ++mt)
    #pragma unroll
    for (int nt = 0; nt < 2; ++nt){
      const int k = colBase + nt*16 + nl;
      #pragma unroll
      for (int r = 0; r < 8; ++r){
        float v = acc[mt][nt][r];
        if (do_silu) v = gsm_silu(v);
        buf[gsm_a_off(mt*16 + r + mhi*8, k)] = (gbf)v;
      }
    }
}

extern "C" __global__ __launch_bounds__(256, 1) void gsm_edge_kernel(
    const float* __restrict__ x, const int* __restrict__ eidx,
    const float* __restrict__ efeat, const float* __restrict__ nlen, const float* __restrict__ nang,
    const gbf* __restrict__ wp,
    const float* __restrict__ b_edge, const float* __restrict__ b_len, const float* __restrict__ b_ang,
    const float* __restrict__ b_up1, const float* __restrict__ b_up2, const float* __restrict__ b_gate,
    const float* __restrict__ ln_g, const float* __restrict__ ln_b,
    const float* __restrict__ b_f1, const float* __restrict__ b_f2,
    const float* __restrict__ b_m1, const float* __restrict__ b_m2,
    float* __restrict__ zbuf, float* __restrict__ mbuf,
    float* __restrict__ gsum, float* __restrict__ gsq, long E)
{
  extern __shared__ char gsm_smem[];
  gbf*   B0  = (gbf*)gsm_smem;          // 4 bf16 activation buffers, A-layout, 32KB each
  gbf*   B1  = B0 + 16384;
  gbf*   B2  = B1 + 16384;
  gbf*   B3  = B2 + 16384;
  float* sF0 = (float*)(B3 + 16384);    // [64][256] f32
  float* sF1 = sF0 + TMB*HD;            // [64][256] f32
  float* sMu = sF1 + TMB*HD;            // [64]
  float* sRs = sMu + TMB;               // [64]

  const int tid     = threadIdx.x;
  const int lane    = tid & 31;
  const int colBase = (tid >> 5) * 32;  // 8 waves x 32 output columns
  const long e0     = (long)blockIdx.x * TMB;

  // ---- stage (vectorized x4): edge_features -> B0, mean(nei_len) -> B1, mean(nei_angle) -> B2
  for (int idx = tid; idx < TMB*HD/4; idx += 256){
    const int m = idx >> 6, k = (idx & 63) * 4;
    const long e = e0 + m;
    v4f v0 = {0.f,0.f,0.f,0.f}, v1 = v0, v2 = v0;
    if (e < E){
      v0 = *(const v4f*)(efeat + e*HD + k);
      const float* pl = nlen + e*3*HD + k;
      v1 = (*(const v4f*)(pl) + *(const v4f*)(pl + HD) + *(const v4f*)(pl + 2*HD)) * (1.0f/3.0f);
      const float* pa = nang + e*3*HD + k;
      v2 = (*(const v4f*)(pa) + *(const v4f*)(pa + HD) + *(const v4f*)(pa + 2*HD)) * (1.0f/3.0f);
    }
    v4bf w0, w1, w2;
    #pragma unroll
    for (int j = 0; j < 4; ++j){ w0[j] = (gbf)v0[j]; w1[j] = (gbf)v1[j]; w2[j] = (gbf)v2[j]; }
    const int o = gsm_a_off(m, k);
    *(v4bf*)(B0 + o) = w0; *(v4bf*)(B1 + o) = w1; *(v4bf*)(B2 + o) = w2;
  }
  __syncthreads();

  v8f acc[4][2];

  // ef = edge_features @ W_edge + b   -> sF0 (f32) and B3 (bf16)
  gsm_init_acc(acc, b_edge, colBase, lane);
  { const gbf* s[1] = {B0}; gsm_gemm(acc, s, 1, KT_EDGE, wp, lane, colBase); }
  gsm_store_f32(acc, sF0, lane, colBase);
  gsm_store_bf16(acc, B3, lane, colBase, 0);
  __syncthreads();

  // lf -> B0
  gsm_init_acc(acc, b_len, colBase, lane);
  { const gbf* s[1] = {B1}; gsm_gemm(acc, s, 1, KT_LEN, wp, lane, colBase); }
  gsm_store_bf16(acc, B0, lane, colBase, 0);
  __syncthreads();

  // af -> B1
  gsm_init_acc(acc, b_ang, colBase, lane);
  { const gbf* s[1] = {B2}; gsm_gemm(acc, s, 1, KT_ANG, wp, lane, colBase); }
  gsm_store_bf16(acc, B1, lane, colBase, 0);
  __syncthreads();

  // h1 = silu([ef,lf,af] @ up1 + b) -> B2
  gsm_init_acc(acc, b_up1, colBase, lane);
  { const gbf* s[3] = {B3, B0, B1}; gsm_gemm(acc, s, 3, KT_UP1, wp, lane, colBase); }
  gsm_store_bf16(acc, B2, lane, colBase, 1);
  __syncthreads();

  // update = h1 @ up2 + b -> sF1 (f32) and B0 (bf16)
  gsm_init_acc(acc, b_up2, colBase, lane);
  { const gbf* s[1] = {B2}; gsm_gemm(acc, s, 1, KT_UP2, wp, lane, colBase); }
  gsm_store_f32(acc, sF1, lane, colBase);
  gsm_store_bf16(acc, B0, lane, colBase, 0);
  __syncthreads();

  // gate = sigmoid([ef,update] @ gate_w + b); t = ef + gate*update -> sF1
  gsm_init_acc(acc, b_gate, colBase, lane);
  { const gbf* s[2] = {B3, B0}; gsm_gemm(acc, s, 2, KT_GATE, wp, lane, colBase); }
  {
    const int mhi = (lane >> 4) & 1, nl = lane & 15;
    #pragma unroll
    for (int mt = 0; mt < 4; ++mt)
      #pragma unroll
      for (int nt = 0; nt < 2; ++nt){
        const int col = colBase + nt*16 + nl;
        #pragma unroll
        for (int r = 0; r < 8; ++r){
          const int m = mt*16 + r + mhi*8;
          const float g = gsm_sigmoid(acc[mt][nt][r]);
          sF1[m*HD + col] = sF0[m*HD + col] + g * sF1[m*HD + col];
        }
      }
  }
  __syncthreads();

  // LayerNorm row stats over t
  if (tid < TMB){
    const float* row = sF1 + tid*HD;
    float s = 0.f, q = 0.f;
    for (int c = 0; c < HD; ++c){ const float v = row[c]; s += v; q += v*v; }
    const float mu = s * (1.0f/HD);
    sMu[tid] = mu;
    sRs[tid] = rsqrtf(q * (1.0f/HD) - mu*mu + 1e-5f);
  }
  __syncthreads();

  // eo = relu(LN(t)) -> B1 ; stage x[dst] -> B0, x[src] -> B2  (vectorized x4)
  for (int idx = tid; idx < TMB*HD/4; idx += 256){
    const int m = idx >> 6, c = (idx & 63) * 4;
    const v4f t  = *(const v4f*)(sF1 + m*HD + c);
    const v4f lg = *(const v4f*)(ln_g + c);
    const v4f lb = *(const v4f*)(ln_b + c);
    const float mu = sMu[m], rs = sRs[m];
    const long e = e0 + m;
    v4f xd = {0.f,0.f,0.f,0.f}, xs = xd;
    if (e < E){
      xd = *(const v4f*)(x + (long)eidx[E + e]*HD + c);   // x_i = dst
      xs = *(const v4f*)(x + (long)eidx[e]*HD + c);       // x_j = src
    }
    v4bf weo, wxd, wxs;
    #pragma unroll
    for (int j = 0; j < 4; ++j){
      weo[j] = (gbf)fmaxf((t[j] - mu) * rs * lg[j] + lb[j], 0.f);
      wxd[j] = (gbf)xd[j];
      wxs[j] = (gbf)xs[j];
    }
    const int o = gsm_a_off(m, c);
    *(v4bf*)(B1 + o) = weo; *(v4bf*)(B0 + o) = wxd; *(v4bf*)(B2 + o) = wxs;
  }
  __syncthreads();

  // t1 = silu([xd,xs,eo] @ full1 + b) -> B3
  gsm_init_acc(acc, b_f1, colBase, lane);
  { const gbf* s[3] = {B0, B2, B1}; gsm_gemm(acc, s, 3, KT_F1, wp, lane, colBase); }
  gsm_store_bf16(acc, B3, lane, colBase, 1);
  __syncthreads();

  // z = t1 @ full2 + b -> sF0; spill + per-channel BN statistics
  gsm_init_acc(acc, b_f2, colBase, lane);
  { const gbf* s[1] = {B3}; gsm_gemm(acc, s, 1, KT_F2, wp, lane, colBase); }
  gsm_store_f32(acc, sF0, lane, colBase);
  __syncthreads();

  for (int idx = tid; idx < TMB*HD/4; idx += 256){
    const int m = idx >> 6;
    if (e0 + m < E)
      *(v4f*)(zbuf + (e0 + m)*HD + (idx & 63)*4) = *(const v4f*)(sF0 + idx*4);
  }
  {
    const int c = tid;                    // blockDim.x == HD
    float s = 0.f, q = 0.f;
    for (int m = 0; m < TMB; ++m)
      if (e0 + m < E){ const float v = sF0[m*HD + c]; s += v; q += v*v; }
    atomicAdd(&gsum[c], s);
    atomicAdd(&gsq[c], q);
  }

  // t2 = silu([xd,xs,eo] @ msg1 + b) -> B3
  gsm_init_acc(acc, b_m1, colBase, lane);
  { const gbf* s[3] = {B0, B2, B1}; gsm_gemm(acc, s, 3, KT_M1, wp, lane, colBase); }
  gsm_store_bf16(acc, B3, lane, colBase, 1);
  __syncthreads();

  // m = t2 @ msg2 + b -> sF0 -> mbuf
  gsm_init_acc(acc, b_m2, colBase, lane);
  { const gbf* s[1] = {B3}; gsm_gemm(acc, s, 1, KT_M2, wp, lane, colBase); }
  gsm_store_f32(acc, sF0, lane, colBase);
  __syncthreads();

  for (int idx = tid; idx < TMB*HD/4; idx += 256){
    const int m = idx >> 6;
    if (e0 + m < E)
      *(v4f*)(mbuf + (e0 + m)*HD + (idx & 63)*4) = *(const v4f*)(sF0 + idx*4);
  }
}

// f32 -> bf16 weight pack into WMMA B-layout: 32x16 blocks, lane = n | (((k>>3)&1)<<4), elem = (k&7) | ((k&16)>>1)
extern "C" __global__ void gsm_pack_kernel(const float* __restrict__ w, int rows, int ktgBase,
                                           gbf* __restrict__ wp){
  for (long i = (long)blockIdx.x*blockDim.x + threadIdx.x; i < (long)rows*HD;
       i += (long)gridDim.x*blockDim.x){
    const int r = (int)(i >> 8), c = (int)(i & (HD-1));
    const int ktg = ktgBase + (r >> 5);
    const int k32 = r & 31;
    const int lane = (c & 15) | (((k32 >> 3) & 1) << 4);
    wp[(size_t)(ktg*16 + (c >> 4))*512 + lane*16 + ((k32 & 7) | ((k32 & 16) >> 1))] = (gbf)w[i];
  }
}

extern "C" __global__ void gsm_zero_kernel(float* __restrict__ p, long n){
  for (long i = (long)blockIdx.x*blockDim.x + threadIdx.x; i < n; i += (long)gridDim.x*blockDim.x)
    p[i] = 0.f;
}

extern "C" __global__ void gsm_scatter_kernel(const float* __restrict__ zbuf, const float* __restrict__ mbuf,
    const float* __restrict__ elen, const int* __restrict__ eidx,
    const float* __restrict__ gsum, const float* __restrict__ gsq,
    const float* __restrict__ bng, const float* __restrict__ bnb,
    float* __restrict__ agg, long E){
  const float invE = 1.0f / (float)E;
  for (long i = (long)blockIdx.x*blockDim.x + threadIdx.x; i < E*(long)HD;
       i += (long)gridDim.x*blockDim.x){
    const long e = i >> 8; const int c = (int)(i & (HD-1));
    const float mean = gsum[c] * invE;
    const float rstd = rsqrtf(gsq[c]*invE - mean*mean + 1e-5f);
    const float score = gsm_sigmoid((zbuf[i] - mean) * rstd * bng[c] + bnb[c]);
    const float el = elen[e];
    float env = 0.f;
    if (el < CUTOFF_F){ const float cv = cosf(el * (M_PI_F / (2.0f * CUTOFF_F))); env = cv*cv; }
    atomicAdd(&agg[(long)eidx[E + e]*HD + c], env * score * mbuf[i]);
  }
}

extern "C" __global__ void gsm_nstat_kernel(const float* __restrict__ agg,
    float* __restrict__ nsum, float* __restrict__ nsq, long N){
  const int c = threadIdx.x;
  const long r0 = (long)blockIdx.x * 64;
  float s = 0.f, q = 0.f;
  for (int i = 0; i < 64; ++i){
    const long n = r0 + i;
    if (n < N){ const float v = agg[n*HD + c]; s += v; q += v*v; }
  }
  atomicAdd(&nsum[c], s);
  atomicAdd(&nsq[c], q);
}

extern "C" __global__ void gsm_final_kernel(const float* __restrict__ x, const float* __restrict__ agg,
    const float* __restrict__ nsum, const float* __restrict__ nsq,
    const float* __restrict__ g, const float* __restrict__ b, float* __restrict__ out, long N){
  const float invN = 1.0f / (float)N;
  for (long i = (long)blockIdx.x*blockDim.x + threadIdx.x; i < N*(long)HD;
       i += (long)gridDim.x*blockDim.x){
    const int c = (int)(i & (HD-1));
    const float mean = nsum[c] * invN;
    const float rstd = rsqrtf(nsq[c]*invN - mean*mean + 1e-5f);
    out[i] = fmaxf(x[i] + (agg[i] - mean) * rstd * g[c] + b[c], 0.f);
  }
}

extern "C" void kernel_launch(void* const* d_in, const int* in_sizes, int n_in,
                              void* d_out, int out_size, void* d_ws, size_t ws_size,
                              hipStream_t stream){
  const float* x      = (const float*)d_in[0];
  const int*   eidx   = (const int*)  d_in[1];
  const float* efeat  = (const float*)d_in[2];
  const float* nlen   = (const float*)d_in[3];
  const float* nang   = (const float*)d_in[4];
  const float* elen   = (const float*)d_in[5];
  const float* w_edge = (const float*)d_in[6];  const float* b_edge = (const float*)d_in[7];
  const float* w_len  = (const float*)d_in[8];  const float* b_len  = (const float*)d_in[9];
  const float* w_ang  = (const float*)d_in[10]; const float* b_ang  = (const float*)d_in[11];
  const float* w_up1  = (const float*)d_in[12]; const float* b_up1  = (const float*)d_in[13];
  const float* w_up2  = (const float*)d_in[14]; const float* b_up2  = (const float*)d_in[15];
  const float* w_gate = (const float*)d_in[16]; const float* b_gate = (const float*)d_in[17];
  const float* ln_g   = (const float*)d_in[18]; const float* ln_b   = (const float*)d_in[19];
  const float* w_f1   = (const float*)d_in[20]; const float* b_f1   = (const float*)d_in[21];
  const float* w_f2   = (const float*)d_in[22]; const float* b_f2   = (const float*)d_in[23];
  const float* w_m1   = (const float*)d_in[24]; const float* b_m1   = (const float*)d_in[25];
  const float* w_m2   = (const float*)d_in[26]; const float* b_m2   = (const float*)d_in[27];
  const float* bn_ig  = (const float*)d_in[28]; const float* bn_ib  = (const float*)d_in[29];
  const float* bn_og  = (const float*)d_in[30]; const float* bn_ob  = (const float*)d_in[31];
  (void)n_in; (void)out_size; (void)ws_size;

  const long N = in_sizes[0] / HD;
  const long E = in_sizes[2] / HD;

  char* ws = (char*)d_ws;
  size_t off = 0;
  auto take = [&](size_t bytes) -> void* {
    off = (off + 255) & ~(size_t)255;
    void* p = ws + off;
    off += bytes;
    return p;
  };
  gbf*   wp   = (gbf*)  take((size_t)KT_TOT * 16 * 512 * sizeof(gbf));
  float* gsum = (float*)take(HD * sizeof(float));   // these five are contiguous (1KB blocks,
  float* gsq  = (float*)take(HD * sizeof(float));   //  256B aligned) -> zeroed in one shot
  float* nsum = (float*)take(HD * sizeof(float));
  float* nsq  = (float*)take(HD * sizeof(float));
  float* agg  = (float*)take((size_t)N * HD * sizeof(float));
  float* zbuf = (float*)take((size_t)E * HD * sizeof(float));
  float* mbuf = (float*)take((size_t)E * HD * sizeof(float));

  gsm_zero_kernel<<<1024, 256, 0, stream>>>(gsum, (long)(4*HD) + N*(long)HD);

  struct P { const float* w; int rows; int base; };
  const P packs[10] = {
    {w_edge, 256, KT_EDGE}, {w_len, 256, KT_LEN}, {w_ang, 256, KT_ANG},
    {w_up1, 768, KT_UP1},   {w_up2, 256, KT_UP2}, {w_gate, 512, KT_GATE},
    {w_f1, 768, KT_F1},     {w_f2, 256, KT_F2},   {w_m1, 768, KT_M1},
    {w_m2, 256, KT_M2}};
  for (int i = 0; i < 10; ++i)
    gsm_pack_kernel<<<256, 256, 0, stream>>>(packs[i].w, packs[i].rows, packs[i].base, wp);

  const size_t smem = 4*(size_t)16384*sizeof(gbf)          // B0..B3
                    + 2*(size_t)TMB*HD*sizeof(float)       // sF0, sF1
                    + 2*(size_t)TMB*sizeof(float);         // sMu, sRs  (= 262,656 B < 320KB)
  gsm_edge_kernel<<<dim3((unsigned)((E + TMB - 1)/TMB)), dim3(256), smem, stream>>>(
      x, eidx, efeat, nlen, nang, wp,
      b_edge, b_len, b_ang, b_up1, b_up2, b_gate, ln_g, ln_b,
      b_f1, b_f2, b_m1, b_m2, zbuf, mbuf, gsum, gsq, E);

  gsm_scatter_kernel<<<4096, 256, 0, stream>>>(zbuf, mbuf, elen, eidx, gsum, gsq,
                                               bn_ig, bn_ib, agg, E);
  gsm_nstat_kernel<<<dim3((unsigned)((N + 63)/64)), dim3(256), 0, stream>>>(agg, nsum, nsq, N);
  gsm_final_kernel<<<2048, 256, 0, stream>>>(x, agg, nsum, nsq, bn_og, bn_ob, (float*)d_out, N);
}
